// QDTrackGraph_26388279067057
// MI455X (gfx1250) — compile-verified, matrix operand order checked by reading, not run on hardware
//
#include <hip/hip_runtime.h>
#include <hip/hip_bf16.h>
#include <math.h>

#define NDET 4096
#define MEMO 8192
#define DIM  256
#define TOPK 16
#define OBJ_THR 0.3f
#define INIT_THR 0.7f
#define MATCH_THR 0.5f
#define NMS_CONF_THR 0.5f
#define NMS_BACKDROP_IOU 0.3f
#define NMS_CLASS_IOU 0.7f
#define NEGV (-1e30f)

typedef __attribute__((ext_vector_type(16))) __bf16 v16bf;
typedef __attribute__((ext_vector_type(8)))  float  v8f;

// ---------------- 1) bitonic sort by (-score, idx) in LDS ----------------
__global__ void qdt_sort_kernel(const float* __restrict__ scores,
                                int* __restrict__ order,
                                float* __restrict__ scores_s) {
    __shared__ float kk[NDET];
    __shared__ int   vv[NDET];
    const int tid = threadIdx.x;
    for (int i = tid; i < NDET; i += 1024) { kk[i] = scores[i]; vv[i] = i; }
    __syncthreads();
    for (int len = 2; len <= NDET; len <<= 1) {
        for (int j = len >> 1; j > 0; j >>= 1) {
            for (int i = tid; i < NDET; i += 1024) {
                int ixj = i ^ j;
                if (ixj > i) {
                    bool dir = ((i & len) == 0);   // ascending in precede-order
                    float ka = kk[i], kb = kk[ixj];
                    int va = vv[i], vb = vv[ixj];
                    bool b_first = (kb > ka) || (kb == ka && vb < va);
                    if (b_first == dir) {
                        kk[i] = kb; kk[ixj] = ka;
                        vv[i] = vb; vv[ixj] = va;
                    }
                }
            }
            __syncthreads();
        }
    }
    for (int i = tid; i < NDET; i += 1024) { order[i] = vv[i]; scores_s[i] = kk[i]; }
}

// ---------------- 2) gather boxes/cls in sorted order ----------------
__global__ void qdt_gather_kernel(const int* __restrict__ order,
                                  const float* __restrict__ det,
                                  const int* __restrict__ cls,
                                  float* __restrict__ boxes_s,
                                  int* __restrict__ cls_s) {
    int i = blockIdx.x * blockDim.x + threadIdx.x;
    if (i >= NDET) return;
    int o = order[i];
    boxes_s[i * 4 + 0] = det[o * 4 + 0];
    boxes_s[i * 4 + 1] = det[o * 4 + 1];
    boxes_s[i * 4 + 2] = det[o * 4 + 2];
    boxes_s[i * 4 + 3] = det[o * 4 + 3];
    cls_s[i] = cls[o];
}

__global__ void qdt_conv_emb_kernel(const int* __restrict__ order,
                                    const float* __restrict__ emb,
                                    __bf16* __restrict__ emb_bf) {
    int idx = blockIdx.x * blockDim.x + threadIdx.x;
    if (idx >= NDET * DIM) return;
    int i = idx >> 8, k = idx & (DIM - 1);
    emb_bf[idx] = (__bf16)emb[order[i] * DIM + k];
}

__global__ void qdt_conv_memo_kernel(const float* __restrict__ memo,
                                     __bf16* __restrict__ memo_bf) {
    int idx = blockIdx.x * blockDim.x + threadIdx.x;
    if (idx >= MEMO * DIM) return;
    memo_bf[idx] = (__bf16)memo[idx];
}

// ---------------- 3) validity mask (dual-threshold parallel NMS) ----------------
__global__ void qdt_valid_kernel(const float* __restrict__ boxes_s,
                                 const float* __restrict__ scores_s,
                                 int* __restrict__ valid) {
    int i = blockIdx.x * blockDim.x + threadIdx.x;
    if (i >= NDET) return;
    float x1 = boxes_s[i * 4 + 0], y1 = boxes_s[i * 4 + 1];
    float x2 = boxes_s[i * 4 + 2], y2 = boxes_s[i * 4 + 3];
    float ai = fmaxf(x2 - x1, 0.f) * fmaxf(y2 - y1, 0.f);
    float thr = (scores_s[i] < OBJ_THR) ? NMS_BACKDROP_IOU : NMS_CLASS_IOU;
    int ok = 1;
    for (int j = 0; j < i; ++j) {
        float bx1 = boxes_s[j * 4 + 0], by1 = boxes_s[j * 4 + 1];
        float bx2 = boxes_s[j * 4 + 2], by2 = boxes_s[j * 4 + 3];
        float aj = fmaxf(bx2 - bx1, 0.f) * fmaxf(by2 - by1, 0.f);
        float lx = fmaxf(x1, bx1), ly = fmaxf(y1, by1);
        float rx = fminf(x2, bx2), ry = fminf(y2, by2);
        float iw = fmaxf(rx - lx, 0.f), ih = fmaxf(ry - ly, 0.f);
        float inter = iw * ih;
        float iou = inter / fmaxf(ai + aj - inter, 1e-6f);
        if (iou > thr) { ok = 0; break; }
    }
    valid[i] = ok;
}

// ---------------- 4) WMMA bf16 GEMM: feats = emb_s @ memoT, masked ----------------
__global__ void qdt_gemm_kernel(const __bf16* __restrict__ A,   // [NDET, DIM] sorted
                                const __bf16* __restrict__ B,   // [MEMO, DIM]
                                const int* __restrict__ valid,
                                float* __restrict__ out) {      // [NDET, MEMO]
    const int wave = threadIdx.x >> 5;
    const int lane = threadIdx.x & 31;
    const int hi = lane >> 4;
    const int m16 = lane & 15;
    const int row0 = blockIdx.y * 64 + wave * 16;
    const int col0 = blockIdx.x * 64;

    v8f acc[4] = {};
    const __bf16* arow = A + (size_t)(row0 + m16) * DIM;

    for (int kb = 0; kb < DIM; kb += 32) {
        v16bf a;
        {
            const __bf16* p0 = arow + kb + hi * 8;
            const __bf16* p1 = p0 + 16;
            #pragma unroll
            for (int t = 0; t < 8; ++t) { a[t] = p0[t]; a[t + 8] = p1[t]; }
        }
        #pragma unroll
        for (int t = 0; t < 4; ++t) {
            const __bf16* brow = B + (size_t)(col0 + t * 16 + m16) * DIM + kb + hi * 8;
            v16bf b;
            #pragma unroll
            for (int u = 0; u < 8; ++u) { b[u] = brow[u]; b[u + 8] = brow[u + 16]; }
            acc[t] = __builtin_amdgcn_wmma_f32_16x16x32_bf16(
                false, a, false, b, (short)0, acc[t], false, false);
        }
    }

    #pragma unroll
    for (int t = 0; t < 4; ++t) {
        int col = col0 + t * 16 + m16;
        #pragma unroll
        for (int r = 0; r < 8; ++r) {
            int row = row0 + r + hi * 8;
            float v = valid[row] ? acc[t][r] : NEGV;
            out[(size_t)row * MEMO + col] = v;
        }
    }
}

// ---------------- 5) online-softmax row stats (block per row) ----------------
__global__ void qdt_rowstats_kernel(const float* __restrict__ feats,
                                    float* __restrict__ rowmax,
                                    float* __restrict__ rowsum) {
    __shared__ float sm[256], ss[256];
    const int row = blockIdx.x, tid = threadIdx.x;
    const float* p = feats + (size_t)row * MEMO;
    float m = -__builtin_inff(), s = 0.f;
    for (int c = tid; c < MEMO; c += 256) {
        float x = p[c];
        float nm = fmaxf(m, x);
        s = s * __expf(m - nm) + __expf(x - nm);
        m = nm;
    }
    sm[tid] = m; ss[tid] = s; __syncthreads();
    for (int off = 128; off; off >>= 1) {
        if (tid < off) {
            float m2 = sm[tid + off], s2 = ss[tid + off];
            float nm = fmaxf(sm[tid], m2);
            ss[tid] = ss[tid] * __expf(sm[tid] - nm) + s2 * __expf(m2 - nm);
            sm[tid] = nm;
        }
        __syncthreads();
    }
    if (tid == 0) { rowmax[row] = sm[0]; rowsum[row] = ss[0]; }
}

// ---------------- 6) column stats: 16 row-chunk partials + combine ----------------
__global__ void qdt_colstats_part_kernel(const float* __restrict__ feats,
                                         float* __restrict__ pmax,
                                         float* __restrict__ psum) {
    int col = blockIdx.x * blockDim.x + threadIdx.x;   // gridDim.x = MEMO/256
    int chunk = blockIdx.y;                            // 16 chunks of 256 rows
    int r0 = chunk * (NDET / 16);
    float m = -__builtin_inff(), s = 0.f;
    for (int r = r0; r < r0 + NDET / 16; ++r) {
        float x = feats[(size_t)r * MEMO + col];
        float nm = fmaxf(m, x);
        s = s * __expf(m - nm) + __expf(x - nm);
        m = nm;
    }
    pmax[(size_t)chunk * MEMO + col] = m;
    psum[(size_t)chunk * MEMO + col] = s;
}

__global__ void qdt_colstats_combine_kernel(const float* __restrict__ pmax,
                                            const float* __restrict__ psum,
                                            float* __restrict__ colmax,
                                            float* __restrict__ colsum) {
    int col = blockIdx.x * blockDim.x + threadIdx.x;
    float m = -__builtin_inff(), s = 0.f;
    #pragma unroll
    for (int c = 0; c < 16; ++c) {
        float m2 = pmax[(size_t)c * MEMO + col];
        float s2 = psum[(size_t)c * MEMO + col];
        float nm = fmaxf(m, m2);
        s = s * __expf(m - nm) + s2 * __expf(m2 - nm);
        m = nm;
    }
    colmax[col] = m; colsum[col] = s;
}

// ---------------- 7) in-place sims = 0.5*(d2t+t2d)*cat_same*valid ----------------
__global__ void qdt_sims_kernel(float* __restrict__ sims,
                                const float* __restrict__ rowmax,
                                const float* __restrict__ rowsum,
                                const float* __restrict__ colmax,
                                const float* __restrict__ colsum,
                                const int* __restrict__ cls_s,
                                const int* __restrict__ memo_cls,
                                const int* __restrict__ valid) {
    const size_t total = (size_t)NDET * MEMO;
    const size_t stride = (size_t)gridDim.x * blockDim.x;
    for (size_t idx = (size_t)blockIdx.x * blockDim.x + threadIdx.x; idx < total; idx += stride) {
        int i = (int)(idx >> 13);
        int j = (int)(idx & (MEMO - 1));
        float x = sims[idx];
        float d2t = __expf(x - rowmax[i]) / rowsum[i];
        float t2d = __expf(x - colmax[j]) / colsum[j];
        float v = 0.5f * (d2t + t2d);
        v = (cls_s[i] == memo_cls[j] && valid[i]) ? v : 0.f;
        sims[idx] = v;
    }
}

// ---------------- 8a) per-row top-16 candidates (value desc, col asc) ----------------
__device__ __forceinline__ bool qdt_better(float v, int j, float v2, int j2) {
    return (v > v2) || (v == v2 && j < j2);
}

__global__ void qdt_topk_kernel(const float* __restrict__ sims,
                                float* __restrict__ topv,
                                int* __restrict__ topj) {
    __shared__ float Lv[TOPK * 256];
    __shared__ int   Lj[TOPK * 256];
    __shared__ int   heads[256];
    const int tid = threadIdx.x;
    const int row = blockIdx.x;

    #pragma unroll
    for (int r = 0; r < TOPK; ++r) { Lv[r * 256 + tid] = -1.f; Lj[r * 256 + tid] = MEMO; }
    heads[tid] = 0;

    const float4* rowv = (const float4*)(sims + (size_t)row * MEMO);
    #pragma unroll
    for (int t = 0; t < 8; ++t) {
        int q = tid + t * 256;
        float4 f = rowv[q];
        int j0 = q * 4;
        float e[4] = {f.x, f.y, f.z, f.w};
        #pragma unroll
        for (int k = 0; k < 4; ++k) {
            float v = e[k]; int j = j0 + k;
            float wv = Lv[(TOPK - 1) * 256 + tid];
            int   wj = Lj[(TOPK - 1) * 256 + tid];
            if (qdt_better(v, j, wv, wj)) {
                int r = TOPK - 1;
                while (r > 0) {
                    float pv = Lv[(r - 1) * 256 + tid];
                    int   pj = Lj[(r - 1) * 256 + tid];
                    if (qdt_better(v, j, pv, pj)) {
                        Lv[r * 256 + tid] = pv; Lj[r * 256 + tid] = pj; --r;
                    } else break;
                }
                Lv[r * 256 + tid] = v; Lj[r * 256 + tid] = j;
            }
        }
    }
    __syncthreads();

    // single-wave 256-way merge: 16 selection rounds, shfl_xor butterfly per round
    if (tid < 32) {
        for (int r = 0; r < TOPK; ++r) {
            float bv = -2.f; int bj = MEMO; int bl = 0;
            #pragma unroll
            for (int q = 0; q < 8; ++q) {
                int lst = tid * 8 + q;
                int h = heads[lst];
                if (h < TOPK) {
                    float v = Lv[h * 256 + lst]; int j = Lj[h * 256 + lst];
                    if (qdt_better(v, j, bv, bj)) { bv = v; bj = j; bl = lst; }
                }
            }
            #pragma unroll
            for (int off = 16; off; off >>= 1) {
                float ov = __shfl_xor(bv, off, 32);
                int   oj = __shfl_xor(bj, off, 32);
                int   ol = __shfl_xor(bl, off, 32);
                if (qdt_better(ov, oj, bv, bj)) { bv = ov; bj = oj; bl = ol; }
            }
            if (tid == 0) { topv[row * TOPK + r] = bv; topj[row * TOPK + r] = bj; }
            if (tid == (bl >> 3)) heads[bl]++;
        }
    }
}

// ---------------- 8b) sequential greedy matching: single wave32 ----------------
// argmax over non-taken columns == first non-taken entry of the row's top-16
// (conf==0 index mismatches are harmless: `ind` is only consumed when conf>0.5).
__global__ void qdt_match_kernel(const float* __restrict__ sims,
                                 const float* __restrict__ topv,
                                 const int* __restrict__ topj,
                                 const float* __restrict__ scores_s,
                                 const int* __restrict__ memo_track,
                                 int* __restrict__ ids_s) {
    __shared__ unsigned tak[MEMO / 32];
    const int lane = threadIdx.x;
    for (int w = lane; w < MEMO / 32; w += 32) tak[w] = 0u;
    __syncthreads();

    for (int i = 0; i < NDET; ++i) {
        float v = -1.f; int j = MEMO; bool cand = false;
        if (lane < TOPK) {
            v = topv[i * TOPK + lane];
            j = topj[i * TOPK + lane];
            cand = (j < MEMO) && !((tak[j >> 5] >> (j & 31)) & 1u);
        }
        unsigned long long m = __ballot(cand);
        float conf; int ind;
        if (m != 0ull) {
            int first = __ffsll(m) - 1;
            conf = __shfl(v, first, 32);
            ind  = __shfl(j, first, 32);
        } else {
            // rare fallback: >16 of this row's best columns taken -> full scan
            float bv = -1.f; int bj = MEMO;
            const float* row = sims + (size_t)i * MEMO;
            for (int c = lane; c < MEMO; c += 32) {
                float x = ((tak[c >> 5] >> (c & 31)) & 1u) ? 0.f : row[c];
                if (x > bv || (x == bv && c < bj)) { bv = x; bj = c; }
            }
            #pragma unroll
            for (int off = 16; off; off >>= 1) {
                float ov = __shfl_xor(bv, off, 32);
                int   oj = __shfl_xor(bj, off, 32);
                if (ov > bv || (ov == bv && oj < bj)) { bv = ov; bj = oj; }
            }
            conf = bv; ind = bj;
        }
        if (lane == 0) {
            int cur = memo_track[ind];
            bool match = conf > MATCH_THR;
            bool is_track = cur > -1;
            bool strong = scores_s[i] > OBJ_THR;
            bool assign = match && is_track && strong;
            ids_s[i] = assign ? cur
                              : ((match && !is_track && conf > NMS_CONF_THR) ? -2 : -1);
            if (assign) tak[ind >> 5] |= (1u << (ind & 31));
        }
        __syncthreads();   // single-wave workgroup: lowers to S_NOP per ISA
    }
}

// ---------------- 9) new-track cumsum + un-permute ----------------
__global__ void qdt_finalize_kernel(const int* __restrict__ ids_s,
                                    const float* __restrict__ scores_s,
                                    const int* __restrict__ valid,
                                    const int* __restrict__ order,
                                    const int* __restrict__ num_tracks_p,
                                    float* __restrict__ out_ids,
                                    float* __restrict__ out_valid) {
    __shared__ int part[1024];
    const int tid = threadIdx.x;
    const int base = tid * 4;
    int nw[4]; int loc = 0;
    #pragma unroll
    for (int t = 0; t < 4; ++t) {
        int i = base + t;
        nw[t] = (ids_s[i] == -1 && scores_s[i] > INIT_THR && valid[i]) ? 1 : 0;
        loc += nw[t];
    }
    part[tid] = loc;
    __syncthreads();
    for (int off = 1; off < 1024; off <<= 1) {
        int add = (tid >= off) ? part[tid - off] : 0;
        __syncthreads();
        part[tid] += add;
        __syncthreads();
    }
    int run = part[tid] - loc;   // exclusive prefix
    int ntr = num_tracks_p[0];
    #pragma unroll
    for (int t = 0; t < 4; ++t) {
        int i = base + t;
        int id = ids_s[i];
        if (nw[t]) { id = ntr + run; run++; }
        int o = order[i];
        out_ids[o] = (float)id;
        out_valid[o] = valid[i] ? 1.f : 0.f;
    }
}

extern "C" void kernel_launch(void* const* d_in, const int* in_sizes, int n_in,
                              void* d_out, int out_size, void* d_ws, size_t ws_size,
                              hipStream_t stream) {
    const float* detections = (const float*)d_in[0];
    const float* scores     = (const float*)d_in[1];
    const int*   cls        = (const int*)d_in[2];
    const float* emb        = (const float*)d_in[3];
    const float* memo_emb   = (const float*)d_in[4];
    const int*   memo_cls   = (const int*)d_in[5];
    const int*   memo_track = (const int*)d_in[6];
    const int*   num_tracks = (const int*)d_in[7];

    float* out       = (float*)d_out;
    float* out_ids   = out;                 // [NDET]
    float* out_valid = out + NDET;          // [NDET]
    float* simsbuf   = out + 2 * NDET;      // [NDET*MEMO] (staged feats -> sims)

    char* w = (char*)d_ws;
    int*    order    = (int*)w;    w += NDET * sizeof(int);
    float*  scores_s = (float*)w;  w += NDET * sizeof(float);
    float*  boxes_s  = (float*)w;  w += NDET * 4 * sizeof(float);
    int*    cls_s    = (int*)w;    w += NDET * sizeof(int);
    int*    valid    = (int*)w;    w += NDET * sizeof(int);
    int*    ids_s    = (int*)w;    w += NDET * sizeof(int);
    float*  rowmax   = (float*)w;  w += NDET * sizeof(float);
    float*  rowsum   = (float*)w;  w += NDET * sizeof(float);
    float*  colmax   = (float*)w;  w += MEMO * sizeof(float);
    float*  colsum   = (float*)w;  w += MEMO * sizeof(float);
    float*  pmax     = (float*)w;  w += (size_t)16 * MEMO * sizeof(float);
    float*  psum     = (float*)w;  w += (size_t)16 * MEMO * sizeof(float);
    float*  topv     = (float*)w;  w += (size_t)NDET * TOPK * sizeof(float);
    int*    topj     = (int*)w;    w += (size_t)NDET * TOPK * sizeof(int);
    __bf16* emb_bf   = (__bf16*)w; w += (size_t)NDET * DIM * sizeof(__bf16);
    __bf16* memo_bf  = (__bf16*)w; w += (size_t)MEMO * DIM * sizeof(__bf16);

    qdt_sort_kernel<<<1, 1024, 0, stream>>>(scores, order, scores_s);
    qdt_gather_kernel<<<(NDET + 255) / 256, 256, 0, stream>>>(order, detections, cls, boxes_s, cls_s);
    qdt_conv_emb_kernel<<<(NDET * DIM + 255) / 256, 256, 0, stream>>>(order, emb, emb_bf);
    qdt_conv_memo_kernel<<<(MEMO * DIM + 255) / 256, 256, 0, stream>>>(memo_emb, memo_bf);
    qdt_valid_kernel<<<(NDET + 127) / 128, 128, 0, stream>>>(boxes_s, scores_s, valid);

    dim3 ggrid(MEMO / 64, NDET / 64);
    qdt_gemm_kernel<<<ggrid, 128, 0, stream>>>(emb_bf, memo_bf, valid, simsbuf);

    qdt_rowstats_kernel<<<NDET, 256, 0, stream>>>(simsbuf, rowmax, rowsum);
    dim3 cgrid(MEMO / 256, 16);
    qdt_colstats_part_kernel<<<cgrid, 256, 0, stream>>>(simsbuf, pmax, psum);
    qdt_colstats_combine_kernel<<<MEMO / 256, 256, 0, stream>>>(pmax, psum, colmax, colsum);
    qdt_sims_kernel<<<8192, 256, 0, stream>>>(simsbuf, rowmax, rowsum, colmax, colsum,
                                              cls_s, memo_cls, valid);

    qdt_topk_kernel<<<NDET, 256, 0, stream>>>(simsbuf, topv, topj);
    qdt_match_kernel<<<1, 32, 0, stream>>>(simsbuf, topv, topj, scores_s, memo_track, ids_s);
    qdt_finalize_kernel<<<1, 1024, 0, stream>>>(ids_s, scores_s, valid, order, num_tracks,
                                                out_ids, out_valid);
}
